// PointerGenerator_30391188587360
// MI455X (gfx1250) — matrix operand order repaired and checked by reading, not compiled, not run on hardware
//
#include <hip/hip_runtime.h>
#include <math.h>

#define VSZ  32000
#define EDIM 128
#define HDIM 256
#define BSZ  16
#define SLEN 400
#define TLEN 50
#define NVBLK 125   // 32000 / 256 cols per vocab block

typedef __bf16 bf16;
typedef __attribute__((ext_vector_type(16))) __bf16 v16bf;
typedef __attribute__((ext_vector_type(8)))  __bf16 v8bf;
typedef __attribute__((ext_vector_type(8)))  float  v8f;
typedef __attribute__((ext_vector_type(4))) unsigned int v4u;
typedef __attribute__((ext_vector_type(8))) int v8i32;

__device__ __forceinline__ float sigf(float x) { return 1.f / (1.f + expf(-x)); }

union FragU { v16bf v; v8bf h[2]; };

// A fragment: 16x32 bf16 tile, row-major source, lane l: m=l&15, half=l>>4
// element e<8  -> k = half*8 + e        (contiguous 8 bf16)
// element e>=8 -> k = 16 + half*8 + e-8 (contiguous 8 bf16)
__device__ __forceinline__ v16bf frag_a(const bf16* base, int ld) {
  int l = threadIdx.x & 31;
  const bf16* p = base + (l & 15) * ld + (l >> 4) * 8;
  FragU u;
  u.h[0] = *(const v8bf*)(p);
  u.h[1] = *(const v8bf*)(p + 16);
  return u.v;
}

// B fragment: 32x16 bf16 tile from TRANSPOSED weight Wt[N,K]; lane l: n=l&15,
// half=l>>4, element e<8 -> k = half*8 + e, e>=8 -> k = 16 + half*8 + e-8
__device__ __forceinline__ v16bf frag_b(const bf16* base, int ld) {
  int l = threadIdx.x & 31;
  const bf16* p = base + (l & 15) * ld + (l >> 4) * 8;
  FragU u;
  u.h[0] = *(const v8bf*)(p);
  u.h[1] = *(const v8bf*)(p + 16);
  return u.v;
}

__device__ __forceinline__ v8f wmma_bf16(v16bf a, v16bf b, v8f c) {
  return __builtin_amdgcn_wmma_f32_16x16x32_bf16(false, a, false, b, (short)0, c,
                                                 false, false);
}

// ---------------- setup kernels ----------------

__global__ void zero_kernel(float* __restrict__ p, int n) {
  int i = blockIdx.x * blockDim.x + threadIdx.x;
  if (i < n) p[i] = 0.f;
}

// src[K,N] f32 -> dst[N,K] bf16 (transpose + convert)
__global__ void cvt_transpose_kernel(const float* __restrict__ src, bf16* __restrict__ dst,
                                     long long K, long long N) {
  long long i = (long long)blockIdx.x * blockDim.x + threadIdx.x;
  if (i >= K * N) return;
  long long n = i % N, k = i / N;
  dst[n * K + k] = (bf16)src[i];
}

__global__ void cvt_cast_kernel(const float* __restrict__ src, bf16* __restrict__ dst,
                                long long n) {
  long long i = (long long)blockIdx.x * blockDim.x + threadIdx.x;
  if (i < n) dst[i] = (bf16)src[i];
}

// xs[(s*B+b)*E + e] = enc_emb[source_ids[b*S+s]*E + e]  (bf16)
__global__ void embed_gather_kernel(const int* __restrict__ ids, const float* __restrict__ emb,
                                    bf16* __restrict__ xs) {
  int i = blockIdx.x * blockDim.x + threadIdx.x;
  if (i >= SLEN * BSZ * EDIM) return;
  int e = i % EDIM;
  int r = i / EDIM;
  int b = r % BSZ, s = r / BSZ;
  int tok = ids[b * SLEN + s];
  xs[i] = (bf16)emb[tok * EDIM + e];
}

// Generic WMMA GEMM: C[M,N] f32 = A[M,K] bf16 (row-major) * Bt[N,K] bf16 + bias
// grid.x = M/16, grid.y = N/128, block = 256 (8 waves, one 16x16 tile each)
__global__ void gemm_bf16_kernel(const bf16* __restrict__ A, const bf16* __restrict__ Bt,
                                 const float* __restrict__ bias, float* __restrict__ C,
                                 int N, int K) {
  int lane = threadIdx.x & 31, wave = threadIdx.x >> 5;
  int m0 = blockIdx.x * 16;
  int n0 = blockIdx.y * 128 + wave * 16;
  v8f acc = {};
  for (int k0 = 0; k0 < K; k0 += 32) {
    v16bf a = frag_a(A + (long long)m0 * K + k0, K);
    v16bf b = frag_b(Bt + (long long)n0 * K + k0, K);
    acc = wmma_bf16(a, b, acc);
  }
  int n = n0 + (lane & 15);
  float bv = bias ? bias[n] : 0.f;
  int mr = m0 + (lane >> 4) * 8;
#pragma unroll
  for (int r = 0; r < 8; ++r) C[(long long)(mr + r) * N + n] = acc[r] + bv;
}

// ---------------- encoder LSTM scans (2 blocks: fwd/bwd, persistent) ----------------

__global__ void enc_lstm_kernel(const float* __restrict__ xWF, const float* __restrict__ xWB,
                                const bf16* __restrict__ WFt, const bf16* __restrict__ WBt,
                                float* __restrict__ enc_out,   // [B,S,2H]
                                float* __restrict__ hfin,      // [2][B*H]
                                float* __restrict__ cst) {     // [2][B*H] (also final c)
  __shared__ bf16 hB[BSZ * HDIM];     // 8 KB
  __shared__ bf16 gS[BSZ * 4 * HDIM]; // 32 KB
  int dir = blockIdx.x;
  const float* xW = dir ? xWB : xWF;
  const bf16* Wt  = dir ? WBt : WFt;
  float* cp = cst + dir * (BSZ * HDIM);
  float* hf = hfin + dir * (BSZ * HDIM);
  int tid = threadIdx.x, lane = tid & 31, wave = tid >> 5;
  for (int i = tid; i < BSZ * HDIM; i += 256) { cp[i] = 0.f; hB[i] = (bf16)0.f; }
  __syncthreads();
  for (int s = 0; s < SLEN; ++s) {
    int spos = dir ? (SLEN - 1 - s) : s;
    // g[16,1024] = h @ Whh^T (WMMA) + precomputed x@Wih + b
    for (int tt = 0; tt < 8; ++tt) {
      int n0 = wave * 128 + tt * 16;
      v8f acc = {};
#pragma unroll
      for (int k0 = 0; k0 < HDIM; k0 += 32) {
        v16bf a = frag_a((const bf16*)hB + k0, HDIM);
        v16bf b = frag_b(Wt + n0 * HDIM + k0, HDIM);
        acc = wmma_bf16(a, b, acc);
      }
      int n = n0 + (lane & 15);
      int mr = (lane >> 4) * 8;
#pragma unroll
      for (int r = 0; r < 8; ++r)
        gS[(mr + r) * 1024 + n] = (bf16)(acc[r] + xW[(spos * BSZ + mr + r) * 1024 + n]);
    }
    __syncthreads();
    for (int i = tid; i < BSZ * HDIM; i += 256) {
      int bb = i >> 8, j = i & 255;
      const bf16* gb = gS + bb * 1024;
      float gi = (float)gb[j], gf = (float)gb[256 + j];
      float gg = (float)gb[512 + j], go = (float)gb[768 + j];
      float cc = sigf(gf) * cp[i] + sigf(gi) * tanhf(gg);
      float hh = sigf(go) * tanhf(cc);
      cp[i] = cc;
      hB[i] = (bf16)hh;
      enc_out[((long long)bb * SLEN + spos) * 512 + dir * HDIM + j] = hh;
      if (s == SLEN - 1) hf[i] = hh;
    }
    __syncthreads();
  }
}

// h0/c0 = tanh(concat(f,b) @ W + b) — grid 2 (h, c), block 256
__global__ void reduce_state_kernel(const float* __restrict__ hfin, const float* __restrict__ cfin,
                                    const bf16* __restrict__ Ht, const bf16* __restrict__ Ct,
                                    const float* __restrict__ hb, const float* __restrict__ cb,
                                    float* __restrict__ hdec, float* __restrict__ cdec) {
  __shared__ bf16 aS[BSZ * 512]; // 16 KB
  int which = blockIdx.x;
  const float* src = which ? cfin : hfin;
  const bf16* Wt = which ? Ct : Ht;
  const float* bias = which ? cb : hb;
  float* out = which ? cdec : hdec;
  int tid = threadIdx.x, lane = tid & 31, wave = tid >> 5;
  for (int i = tid; i < BSZ * 512; i += 256) {
    int bb = i >> 9, j = i & 511;
    float v = (j < HDIM) ? src[bb * HDIM + j] : src[BSZ * HDIM + bb * HDIM + (j - HDIM)];
    aS[i] = (bf16)v;
  }
  __syncthreads();
  for (int tt = 0; tt < 2; ++tt) {
    int n0 = wave * 32 + tt * 16;
    v8f acc = {};
#pragma unroll
    for (int k0 = 0; k0 < 512; k0 += 32) {
      v16bf a = frag_a((const bf16*)aS + k0, 512);
      v16bf b = frag_b(Wt + n0 * 512 + k0, 512);
      acc = wmma_bf16(a, b, acc);
    }
    int n = n0 + (lane & 15);
    int mr = (lane >> 4) * 8;
#pragma unroll
    for (int r = 0; r < 8; ++r) out[(mr + r) * HDIM + n] = tanhf(acc[r] + bias[n]);
  }
}

// ---------------- decoder step kernels ----------------

// Build A = [emb_t | context | h_prev] bf16 [16,896]; save emb f32 for pgen
__global__ void dec_prep_kernel(const int* __restrict__ din, const float* __restrict__ demb,
                                const float* __restrict__ ctx, const float* __restrict__ hdec,
                                bf16* __restrict__ Adec, float* __restrict__ embt, int t) {
  int i = blockIdx.x * blockDim.x + threadIdx.x;
  if (i >= BSZ * 896) return;
  int bb = i / 896, j = i % 896;
  float v;
  if (j < EDIM) {
    int tok = din[bb * TLEN + t];
    v = demb[tok * EDIM + j];
    embt[bb * EDIM + j] = v;
  } else if (j < EDIM + 512) {
    v = ctx[bb * 512 + (j - EDIM)];
  } else {
    v = hdec[bb * HDIM + (j - EDIM - 512)];
  }
  Adec[i] = (bf16)v;
}

// g = A @ [Wih;Whh]^T + b ; LSTM cell update (1 block, 8 waves)
__global__ void dec_cell_kernel(const bf16* __restrict__ Adec, const bf16* __restrict__ Wiht,
                                const bf16* __restrict__ Whht, const float* __restrict__ bias,
                                float* __restrict__ hdec, float* __restrict__ cdec) {
  __shared__ bf16 gS[BSZ * 4 * HDIM]; // 32 KB
  int tid = threadIdx.x, lane = tid & 31, wave = tid >> 5;
  for (int tt = 0; tt < 8; ++tt) {
    int n0 = wave * 128 + tt * 16;
    v8f acc = {};
    for (int k0 = 0; k0 < 896; k0 += 32) {
      v16bf a = frag_a(Adec + k0, 896);
      v16bf b = (k0 < 640) ? frag_b(Wiht + n0 * 640 + k0, 640)
                           : frag_b(Whht + n0 * HDIM + (k0 - 640), HDIM);
      acc = wmma_bf16(a, b, acc);
    }
    int n = n0 + (lane & 15);
    int mr = (lane >> 4) * 8;
#pragma unroll
    for (int r = 0; r < 8; ++r) gS[(mr + r) * 1024 + n] = (bf16)(acc[r] + bias[n]);
  }
  __syncthreads();
  for (int i = tid; i < BSZ * HDIM; i += 256) {
    int bb = i >> 8, j = i & 255;
    const bf16* gb = gS + bb * 1024;
    float gi = (float)gb[j], gf = (float)gb[256 + j];
    float gg = (float)gb[512 + j], go = (float)gb[768 + j];
    float cc = sigf(gf) * cdec[i] + sigf(gi) * tanhf(gg);
    float hh = sigf(go) * tanhf(cc);
    cdec[i] = cc;
    hdec[i] = hh;
  }
}

// hWs = h @ attn_Ws + b (1 block)
__global__ void hws_kernel(const float* __restrict__ hdec, const bf16* __restrict__ Wst,
                           const float* __restrict__ bias, float* __restrict__ hWs) {
  __shared__ bf16 aS[BSZ * HDIM]; // 8 KB
  int tid = threadIdx.x, lane = tid & 31, wave = tid >> 5;
  for (int i = tid; i < BSZ * HDIM; i += 256) aS[i] = (bf16)hdec[i];
  __syncthreads();
  for (int tt = 0; tt < 2; ++tt) {
    int n0 = wave * 32 + tt * 16;
    v8f acc = {};
#pragma unroll
    for (int k0 = 0; k0 < HDIM; k0 += 32) {
      v16bf a = frag_a((const bf16*)aS + k0, HDIM);
      v16bf b = frag_b(Wst + n0 * HDIM + k0, HDIM);
      acc = wmma_bf16(a, b, acc);
    }
    int n = n0 + (lane & 15);
    int mr = (lane >> 4) * 8;
#pragma unroll
    for (int r = 0; r < 8; ++r) hWs[(mr + r) * HDIM + n] = acc[r] + bias[n];
  }
}

__device__ __forceinline__ float blk_reduce(float v, float* red, bool ismax) {
  int tid = threadIdx.x;
  red[tid] = v;
  __syncthreads();
  for (int off = 128; off > 0; off >>= 1) {
    if (tid < off) red[tid] = ismax ? fmaxf(red[tid], red[tid + off]) : red[tid] + red[tid + off];
    __syncthreads();
  }
  float r = red[0];
  __syncthreads();
  return r;
}

// Attention + coverage + copy-prob + pgen + vocab-A prep. grid B=16, block 256.
__global__ void attn_kernel(const int* __restrict__ src_ids, const int* __restrict__ src_ext,
                            const float* __restrict__ enc_feat, const float* __restrict__ enc_out,
                            const float* __restrict__ hWs, const float* __restrict__ Wc,
                            const float* __restrict__ av, float* __restrict__ coverage,
                            float* __restrict__ ctx, const float* __restrict__ hdec,
                            const float* __restrict__ embt, const float* __restrict__ pgW,
                            const float* __restrict__ pgb0, const float* __restrict__ pgb1,
                            const int* __restrict__ tgt_ext, int t, float* __restrict__ pg_out,
                            float* __restrict__ covloss, float* __restrict__ copyprob,
                            bf16* __restrict__ vocabA) {
  int b = blockIdx.x, tid = threadIdx.x;
  __shared__ float sc[SLEN];
  __shared__ float red[256];
  for (int s = tid; s < SLEN; s += 256) {
    float cov = coverage[b * SLEN + s];
    const float* f = enc_feat + ((long long)b * SLEN + s) * HDIM;
    float acc = 0.f;
    for (int j = 0; j < HDIM; ++j) acc += tanhf(f[j] + hWs[b * HDIM + j] + cov * Wc[j]) * av[j];
    sc[s] = (src_ids[b * SLEN + s] == 0) ? -1e9f : acc;
  }
  __syncthreads();
  float lm = -3e38f;
  for (int s = tid; s < SLEN; s += 256) lm = fmaxf(lm, sc[s]);
  float mx = blk_reduce(lm, red, true);
  float ls = 0.f;
  for (int s = tid; s < SLEN; s += 256) { float e = expf(sc[s] - mx); sc[s] = e; ls += e; }
  float inv = 1.f / blk_reduce(ls, red, false);
  int tg = tgt_ext[b * TLEN + t];
  float lcov = 0.f, lcopy = 0.f;
  for (int s = tid; s < SLEN; s += 256) {
    float a = sc[s] * inv;
    float cv = coverage[b * SLEN + s];
    lcov += fminf(a, cv);
    coverage[b * SLEN + s] = cv + a;
    if (src_ext[b * SLEN + s] == tg) lcopy += a;
    sc[s] = a;
  }
  float cl = blk_reduce(lcov, red, false);
  float cp = blk_reduce(lcopy, red, false);
  if (tid == 0) { covloss[b] = cl; copyprob[b] = cp; }
  __syncthreads();
  for (int d = tid; d < 512; d += 256) {
    float acc = 0.f;
    for (int s = 0; s < SLEN; ++s) acc += sc[s] * enc_out[((long long)b * SLEN + s) * 512 + d];
    ctx[b * 512 + d] = acc;
    vocabA[b * 768 + d] = (bf16)acc;
  }
  for (int d = tid; d < HDIM; d += 256) vocabA[b * 768 + 512 + d] = (bf16)hdec[b * HDIM + d];
  __syncthreads();
  float lp = 0.f;
  for (int j = tid; j < 896; j += 256) {
    float xv = (j < 512) ? ctx[b * 512 + j]
             : (j < 768) ? hdec[b * HDIM + (j - 512)]
                         : embt[b * EDIM + (j - 768)];
    lp += xv * pgW[j];
  }
  float dot = blk_reduce(lp, red, false);
  if (tid == 0) {
    float p = sigf(dot + pgb0[0] + pgb1[0]);
    pg_out[b] = fminf(0.9f, fmaxf(0.1f, p));
  }
}

// Vocab logits via WMMA + streaming softmax partials. grid NVBLK, block 256.
// A operand (16x768 bf16 = 24 KB) is staged into LDS with the Tensor Data
// Mover (tensor_load_to_lds, TENSORcnt), B weights stream from L2 as b128.
__global__ void vocab_gemm_kernel(const bf16* __restrict__ Av, const bf16* __restrict__ Wt,
                                  const float* __restrict__ bias, float* __restrict__ pmax,
                                  float* __restrict__ psum) {
  __shared__ bf16 aS[BSZ * 768];      // 24 KB, filled by TDM
  __shared__ float logits[BSZ * 256]; // 16 KB
  __shared__ float red[256];
  int tid = threadIdx.x, lane = tid & 31, wave = tid >> 5;
  int blk = blockIdx.x;
  if (wave == 0) {
    // Build Tensor DMA Descriptor (D#): 2D tensor, 2-byte elements,
    // tensor/tile = 768 x 16, stride 768, dest = aS.
    unsigned long long ga = (unsigned long long)(size_t)Av;
    unsigned int lds = (unsigned int)(size_t)(&aS[0]); // LDS aperture: addr[31:0] = LDS offset
    v4u g0;
    g0[0] = 1u;                                            // count=1 (valid user D#)
    g0[1] = lds;                                           // lds_addr
    g0[2] = (unsigned int)(ga & 0xffffffffu);              // global_addr[31:0]
    g0[3] = (unsigned int)((ga >> 32) & 0x1ffffffu) | 0x80000000u; // addr[56:32] | type=2
    v8i32 g1;
    g1[0] = (int)(1u << 16);    // workgroup_mask=0, data_size=1 (2 bytes)
    g1[1] = (int)(768u << 16);  // tensor_dim0[15:0] = 768
    g1[2] = (int)(16u << 16);   // tensor_dim0[31:16]=0 | tensor_dim1[15:0]=16
    g1[3] = (int)(768u << 16);  // tensor_dim1[31:16]=0 | tile_dim0=768
    g1[4] = 16;                 // tile_dim1=16, tile_dim2=0
    g1[5] = 768;                // tensor_dim0_stride[31:0]
    g1[6] = (int)(768u << 16);  // stride[47:32]=0 | tensor_dim1_stride[15:0]=768
    g1[7] = 0;                  // tensor_dim1_stride[47:16]=0
    asm volatile("tensor_load_to_lds %0, %1" : : "s"(g0), "s"(g1) : "memory");
    __builtin_amdgcn_s_wait_tensorcnt(0);
  }
  __syncthreads();
  for (int tt = 0; tt < 2; ++tt) {
    int nl0 = wave * 32 + tt * 16;
    int ng0 = blk * 256 + nl0;
    v8f acc = {};
    for (int k0 = 0; k0 < 768; k0 += 32) {
      __builtin_prefetch((const void*)(Wt + (long long)ng0 * 768 + k0 + 512), 0, 1);
      v16bf a = frag_a((const bf16*)aS + k0, 768);
      v16bf b = frag_b(Wt + (long long)ng0 * 768 + k0, 768);
      acc = wmma_bf16(a, b, acc);
    }
    int nl = nl0 + (lane & 15);
    int mr = (lane >> 4) * 8;
#pragma unroll
    for (int r = 0; r < 8; ++r) logits[(mr + r) * 256 + nl] = acc[r] + bias[blk * 256 + nl];
  }
  __syncthreads();
  int row = tid >> 4, sub = tid & 15;
  float lm = -3e38f;
  for (int c = sub; c < 256; c += 16) lm = fmaxf(lm, logits[row * 256 + c]);
  red[tid] = lm;
  __syncthreads();
  for (int off = 8; off > 0; off >>= 1) {
    if (sub < off) red[tid] = fmaxf(red[tid], red[tid + off]);
    __syncthreads();
  }
  float rm = red[row << 4];
  __syncthreads();
  float lsum = 0.f;
  for (int c = sub; c < 256; c += 16) lsum += expf(logits[row * 256 + c] - rm);
  red[tid] = lsum;
  __syncthreads();
  for (int off = 8; off > 0; off >>= 1) {
    if (sub < off) red[tid] += red[tid + off];
    __syncthreads();
  }
  if (sub == 0) { pmax[blk * BSZ + row] = rm; psum[blk * BSZ + row] = red[tid]; }
}

// Combine partials, pull logit at target, accumulate loss. 1 block.
__global__ void dec_final_kernel(const float* __restrict__ pmax, const float* __restrict__ psum,
                                 const float* __restrict__ ctx, const float* __restrict__ hdec,
                                 const float* __restrict__ vocab_W, const float* __restrict__ vocab_b,
                                 const int* __restrict__ tgt_ext, int t,
                                 const float* __restrict__ pg, const float* __restrict__ covloss,
                                 const float* __restrict__ copyprob,
                                 const float* __restrict__ lam, float* __restrict__ acc) {
  int tid = threadIdx.x;
  if (tid >= BSZ) return;
  int b = tid;
  float M = -3e38f;
  for (int i = 0; i < NVBLK; ++i) M = fmaxf(M, pmax[i * BSZ + b]);
  float Ssum = 0.f;
  for (int i = 0; i < NVBLK; ++i) Ssum += psum[i * BSZ + b] * expf(pmax[i * BSZ + b] - M);
  int tg = tgt_ext[b * TLEN + t];
  float vp = 0.f;
  if (tg < VSZ) {
    float lg = vocab_b[tg];
    for (int j = 0; j < 768; ++j) {
      float xv = (j < 512) ? ctx[b * 512 + j] : hdec[b * HDIM + (j - 512)];
      lg += xv * vocab_W[(long long)j * VSZ + tg];
    }
    vp = expf(lg - M) / Ssum;
  }
  float p = pg[b] * vp + (1.f - pg[b]) * copyprob[b];
  float loss = -logf(p + 1e-12f);
  float m = (tg != 0) ? 1.f : 0.f;
  atomicAdd(&acc[0], (loss + lam[0] * covloss[b]) * m);
  atomicAdd(&acc[1], m);
}

__global__ void output_kernel(const float* __restrict__ acc, float* __restrict__ out) {
  out[0] = acc[0] / fmaxf(acc[1], 1.0f);
}

// ---------------- host ----------------

extern "C" void kernel_launch(void* const* d_in, const int* in_sizes, int n_in,
                              void* d_out, int out_size, void* d_ws, size_t ws_size,
                              hipStream_t stream) {
  (void)in_sizes; (void)n_in; (void)out_size; (void)ws_size;
  const int*   source_ids     = (const int*)d_in[0];
  const int*   source_ext_ids = (const int*)d_in[1];
  const int*   decoder_input  = (const int*)d_in[2];
  const int*   target_ext_ids = (const int*)d_in[3];
  const float* enc_emb   = (const float*)d_in[5];
  const float* enc_Wih_f = (const float*)d_in[6];
  const float* enc_Whh_f = (const float*)d_in[7];
  const float* enc_b_f   = (const float*)d_in[8];
  const float* enc_Wih_b = (const float*)d_in[9];
  const float* enc_Whh_b = (const float*)d_in[10];
  const float* enc_b_b   = (const float*)d_in[11];
  const float* red_h_W   = (const float*)d_in[12];
  const float* red_h_b   = (const float*)d_in[13];
  const float* red_c_W   = (const float*)d_in[14];
  const float* red_c_b   = (const float*)d_in[15];
  const float* dec_emb   = (const float*)d_in[16];
  const float* dec_Wih   = (const float*)d_in[17];
  const float* dec_Whh   = (const float*)d_in[18];
  const float* dec_b     = (const float*)d_in[19];
  const float* attn_Wh   = (const float*)d_in[20];
  const float* attn_Ws   = (const float*)d_in[21];
  const float* attn_Ws_b = (const float*)d_in[22];
  const float* attn_Wc   = (const float*)d_in[23];
  const float* attn_v    = (const float*)d_in[24];
  const float* pgen_W    = (const float*)d_in[25];
  const float* pgen_b    = (const float*)d_in[26];
  const float* vocab_W   = (const float*)d_in[27];
  const float* vocab_b   = (const float*)d_in[28];
  const float* lambda_cov = (const float*)d_in[29];
  const float* p_gen_bias = (const float*)d_in[30];

  char* w = (char*)d_ws;
  size_t off = 0;
  auto alloc = [&](size_t bytes) -> char* {
    char* p = w + off;
    off = (off + bytes + 255) & ~(size_t)255;
    return p;
  };
  bf16*  xs_bf   = (bf16*)alloc((size_t)SLEN * BSZ * EDIM * 2);
  float* xWihF   = (float*)alloc((size_t)SLEN * BSZ * 1024 * 4);
  float* xWihB   = (float*)alloc((size_t)SLEN * BSZ * 1024 * 4);
  float* enc_out = (float*)alloc((size_t)BSZ * SLEN * 512 * 4);
  bf16*  enc_out_bf = (bf16*)alloc((size_t)BSZ * SLEN * 512 * 2);
  float* enc_feat   = (float*)alloc((size_t)BSZ * SLEN * HDIM * 4);
  bf16*  WihFt  = (bf16*)alloc((size_t)1024 * EDIM * 2);
  bf16*  WihBt  = (bf16*)alloc((size_t)1024 * EDIM * 2);
  bf16*  WhhFt  = (bf16*)alloc((size_t)1024 * HDIM * 2);
  bf16*  WhhBt  = (bf16*)alloc((size_t)1024 * HDIM * 2);
  bf16*  dWiht  = (bf16*)alloc((size_t)1024 * 640 * 2);
  bf16*  dWhht  = (bf16*)alloc((size_t)1024 * HDIM * 2);
  bf16*  aWht   = (bf16*)alloc((size_t)HDIM * 512 * 2);
  bf16*  aWst   = (bf16*)alloc((size_t)HDIM * HDIM * 2);
  bf16*  rHt    = (bf16*)alloc((size_t)HDIM * 512 * 2);
  bf16*  rCt    = (bf16*)alloc((size_t)HDIM * 512 * 2);
  bf16*  vWt    = (bf16*)alloc((size_t)VSZ * 768 * 2);
  float* hfin = (float*)alloc(2 * BSZ * HDIM * 4);
  float* cst  = (float*)alloc(2 * BSZ * HDIM * 4);
  float* hdec = (float*)alloc(BSZ * HDIM * 4);
  float* cdec = (float*)alloc(BSZ * HDIM * 4);
  float* coverage = (float*)alloc(BSZ * SLEN * 4);
  float* ctx  = (float*)alloc(BSZ * 512 * 4);
  float* hWs  = (float*)alloc(BSZ * HDIM * 4);
  bf16*  Adec = (bf16*)alloc(BSZ * 896 * 2);
  float* embt = (float*)alloc(BSZ * EDIM * 4);
  bf16*  vocabA = (bf16*)alloc(BSZ * 768 * 2);
  float* pg  = (float*)alloc(BSZ * 4);
  float* cvl = (float*)alloc(BSZ * 4);
  float* cpp = (float*)alloc(BSZ * 4);
  float* pmax = (float*)alloc(NVBLK * BSZ * 4);
  float* psum = (float*)alloc(NVBLK * BSZ * 4);
  float* accb = (float*)alloc(2 * 4);

  auto gridFor = [](long long n) { return (unsigned)((n + 255) / 256); };
  // init
  zero_kernel<<<gridFor(BSZ * SLEN), 256, 0, stream>>>(coverage, BSZ * SLEN);
  zero_kernel<<<gridFor(BSZ * 512), 256, 0, stream>>>(ctx, BSZ * 512);
  zero_kernel<<<1, 256, 0, stream>>>(accb, 2);
  // weight convert/transpose to bf16 [N,K]
  auto ct = [&](const float* s, bf16* d, long long K, long long N) {
    cvt_transpose_kernel<<<gridFor(K * N), 256, 0, stream>>>(s, d, K, N);
  };
  ct(enc_Wih_f, WihFt, EDIM, 1024);
  ct(enc_Wih_b, WihBt, EDIM, 1024);
  ct(enc_Whh_f, WhhFt, HDIM, 1024);
  ct(enc_Whh_b, WhhBt, HDIM, 1024);
  ct(dec_Wih, dWiht, 640, 1024);
  ct(dec_Whh, dWhht, HDIM, 1024);
  ct(attn_Wh, aWht, 512, HDIM);
  ct(attn_Ws, aWst, HDIM, HDIM);
  ct(red_h_W, rHt, 512, HDIM);
  ct(red_c_W, rCt, 512, HDIM);
  ct(vocab_W, vWt, 768, VSZ);
  // encoder inputs + x@Wih GEMMs (bias folded here)
  embed_gather_kernel<<<gridFor((long long)SLEN * BSZ * EDIM), 256, 0, stream>>>(
      source_ids, enc_emb, xs_bf);
  dim3 gx(SLEN * BSZ / 16, 1024 / 128);
  gemm_bf16_kernel<<<gx, 256, 0, stream>>>(xs_bf, WihFt, enc_b_f, xWihF, 1024, EDIM);
  gemm_bf16_kernel<<<gx, 256, 0, stream>>>(xs_bf, WihBt, enc_b_b, xWihB, 1024, EDIM);
  // LSTM scans (persistent, WMMA recurrence)
  enc_lstm_kernel<<<2, 256, 0, stream>>>(xWihF, xWihB, WhhFt, WhhBt, enc_out, hfin, cst);
  reduce_state_kernel<<<2, 256, 0, stream>>>(hfin, cst, rHt, rCt, red_h_b, red_c_b, hdec, cdec);
  // enc_feat = enc_out @ attn_Wh
  cvt_cast_kernel<<<gridFor((long long)BSZ * SLEN * 512), 256, 0, stream>>>(
      enc_out, enc_out_bf, (long long)BSZ * SLEN * 512);
  dim3 gf(BSZ * SLEN / 16, HDIM / 128);
  gemm_bf16_kernel<<<gf, 256, 0, stream>>>(enc_out_bf, aWht, nullptr, enc_feat, HDIM, 512);
  // decoder loop
  for (int t = 0; t < TLEN; ++t) {
    dec_prep_kernel<<<gridFor(BSZ * 896), 256, 0, stream>>>(decoder_input, dec_emb, ctx, hdec,
                                                            Adec, embt, t);
    dec_cell_kernel<<<1, 256, 0, stream>>>(Adec, dWiht, dWhht, dec_b, hdec, cdec);
    hws_kernel<<<1, 256, 0, stream>>>(hdec, aWst, attn_Ws_b, hWs);
    attn_kernel<<<BSZ, 256, 0, stream>>>(source_ids, source_ext_ids, enc_feat, enc_out, hWs,
                                         attn_Wc, attn_v, coverage, ctx, hdec, embt, pgen_W,
                                         pgen_b, p_gen_bias, target_ext_ids, t, pg, cvl, cpp,
                                         vocabA);
    vocab_gemm_kernel<<<NVBLK, 256, 0, stream>>>(vocabA, vWt, vocab_b, pmax, psum);
    dec_final_kernel<<<1, 64, 0, stream>>>(pmax, psum, ctx, hdec, vocab_W, vocab_b,
                                           target_ext_ids, t, pg, cvl, cpp, lambda_cov, accb);
  }
  output_kernel<<<1, 1, 0, stream>>>(accb, (float*)d_out);
}
